// TuckerGroupLinear_41755672052502
// MI455X (gfx1250) — matrix operand order repaired and checked by reading, not compiled, not run on hardware
//
#include <hip/hip_runtime.h>
#include <hip/hip_bf16.h>

typedef __attribute__((ext_vector_type(16))) __bf16 bf16x16;
typedef __attribute__((ext_vector_type(8)))  float  f32x8;
typedef __attribute__((ext_vector_type(4)))  unsigned int u32x4;
typedef __attribute__((ext_vector_type(8)))  int i32x8;
typedef __attribute__((ext_vector_type(4)))  int i32x4;

#define N_TOK   8192
#define IN_DIM  4096
#define OUT_DIM 4096
#define NEXP    64
#define DU      256
#define DD      256

#define TILE_K  32
#define BM      64
#define BN      128
// LDS element offsets (unsigned short units)
#define A_BUF_ELEMS (BM * TILE_K)          // 2048 (4 KB)
#define B_BUF_ELEMS (BN * TILE_K)          // 4096 (8 KB)
#define SMEM_ELEMS  (2 * A_BUF_ELEMS + 2 * B_BUF_ELEMS)   // 24 KB

#if __has_builtin(__builtin_amdgcn_tensor_load_to_lds)
#define HAVE_TDM 1
#else
#define HAVE_TDM 0
#endif

union FragAB { bf16x16 v; uint4 q[2]; };

__device__ __forceinline__ f32x8 wmma_bf16(bf16x16 a, bf16x16 b, f32x8 c) {
    return __builtin_amdgcn_wmma_f32_16x16x32_bf16(
        false, a, false, b, (short)0, c, false, false);
}

__device__ __forceinline__ unsigned short f2bf(float f) {
    unsigned int u = __builtin_bit_cast(unsigned int, f);
    u += 0x7FFFu + ((u >> 16) & 1u);          // round-to-nearest-even
    return (unsigned short)(u >> 16);
}

#if HAVE_TDM
// Issue one TDM 2D-tile load: rows x TILE_K bf16 elements, row stride
// `row_stride_elems`, from `gptr` into LDS byte address `lds_byte_addr`.
// D# layout per CDNA5 ISA (group0 128b, group1 256b).
__device__ __forceinline__ void tdm_load_2d(unsigned lds_byte_addr,
                                            const unsigned short* gptr,
                                            int rows, int row_stride_elems) {
    unsigned long long ga = (unsigned long long)(size_t)gptr;
    u32x4 g0;
    g0[0] = 1u;                                        // count=1, user descriptor
    g0[1] = lds_byte_addr;                             // lds_addr
    g0[2] = (unsigned)(ga & 0xFFFFFFFFu);              // global_addr[31:0]
    g0[3] = (unsigned)((ga >> 32) & 0x01FFFFFFu)       // global_addr[56:32]
          | (2u << 30);                                // type=2 ("image")
    i32x8 g1;
    g1[0] = (int)(1u << 16);                           // data_size=1 (2 bytes)
    g1[1] = (int)((unsigned)TILE_K << 16);             // tensor_dim0[15:0] = 32
    g1[2] = (int)((unsigned)rows << 16);               // tensor_dim1[15:0]
    g1[3] = (int)((unsigned)TILE_K << 16);             // tile_dim0 = 32
    g1[4] = rows;                                      // tile_dim1 (tile_dim2=0)
    g1[5] = row_stride_elems;                          // tensor_dim0_stride[31:0]
    g1[6] = 0;                                         // stride hi / dim1_stride
    g1[7] = 0;
    i32x4 gz4 = {0, 0, 0, 0};                          // groups 2/3: dims >2 unused
    i32x8 gz8 = {0, 0, 0, 0, 0, 0, 0, 0};
    __builtin_amdgcn_tensor_load_to_lds(g0, g1, gz4, gz4, gz8, 0);
}
#endif

// ---------------- token -> expert grouping (counting sort) ----------------

__global__ void zero_counts_kernel(unsigned int* counts, unsigned int* cursors) {
    if (threadIdx.x < NEXP) { counts[threadIdx.x] = 0u; cursors[threadIdx.x] = 0u; }
}

__global__ void hist_kernel(const int* __restrict__ eidx, unsigned int* __restrict__ counts) {
    int t = blockIdx.x * blockDim.x + threadIdx.x;
    if (t < N_TOK) atomicAdd(&counts[eidx[t]], 1u);
}

__global__ void prefix_kernel(const unsigned int* __restrict__ counts,
                              unsigned int* __restrict__ offsets,
                              unsigned int* __restrict__ cursors) {
    if (threadIdx.x == 0) {
        unsigned int acc = 0;
        for (int e = 0; e < NEXP; ++e) {
            offsets[e] = acc; cursors[e] = acc; acc += counts[e];
        }
    }
}

__global__ void scatter_kernel(const int* __restrict__ eidx,
                               unsigned int* __restrict__ cursors,
                               int* __restrict__ sorted) {
    int t = blockIdx.x * blockDim.x + threadIdx.x;
    if (t < N_TOK) {
        unsigned int p = atomicAdd(&cursors[eidx[t]], 1u);
        sorted[p] = t;
    }
}

// ---------------- U_in transpose: [IN_DIM, DD] -> [DD, IN_DIM] ----------------

__global__ void transpose_uin_kernel(const unsigned short* __restrict__ U_in,
                                     unsigned short* __restrict__ U_in_T) {
    int idx = blockIdx.x * blockDim.x + threadIdx.x;   // index into output
    int k = idx & (IN_DIM - 1);
    int d = idx >> 12;                                 // IN_DIM == 2^12
    if (d < DD) U_in_T[idx] = U_in[k * DD + d];
}

// ---------------- LDS/TDM double-buffered GEMM: C = A[M,K] * Bs[Nd,K]^T ------
// Block = 8 waves (4M x 2N), block tile 64M x 128N, wave tile 16M x 64N.
// Per K-step: wave 0 DMAs next A(64x32) + B(128x32) tiles into the spare LDS
// buffer while all waves ds_load + WMMA the current buffer. All fragment
// ds_loads are issued before the WMMA burst (one dscnt wait per K-step).

__global__ __launch_bounds__(256)
void gemm_bt_kernel(const unsigned short* __restrict__ A,
                    const unsigned short* __restrict__ Bs,
                    unsigned short* __restrict__ C,
                    int M, int Nd, int K) {
    __shared__ __align__(128) unsigned short smem[SMEM_ELEMS];

    const int lane = threadIdx.x & 31;
    const int wave = threadIdx.x >> 5;
    const int wm = wave & 3, wn = wave >> 2;
    const int hf   = lane >> 4;
    const int lidx = lane & 15;
    const int m_base = blockIdx.x * BM + wm * 16;
    const int n_base = blockIdx.y * BN + wn * 64;

    const unsigned short* Atile = A  + (size_t)(blockIdx.x * BM) * K;
    const unsigned short* Btile = Bs + (size_t)(blockIdx.y * BN) * K;

    f32x8 zero = {0.f, 0.f, 0.f, 0.f, 0.f, 0.f, 0.f, 0.f};
    f32x8 acc[4] = {zero, zero, zero, zero};

    const int KT = K / TILE_K;

#if HAVE_TDM
    // low 32 bits of a generic LDS pointer == LDS byte offset (flat aperture rule)
    const unsigned lds_base = (unsigned)(size_t)(void*)&smem[0];
    if (wave == 0) {
        tdm_load_2d(lds_base,                     Atile, BM, K);
        tdm_load_2d(lds_base + 2u * A_BUF_ELEMS,  Btile, BN, K);
    }
    for (int kt = 0; kt < KT; ++kt) {
        const int b = kt & 1;
        if (wave == 0) __builtin_amdgcn_s_wait_tensorcnt(0);
        __syncthreads();                       // cur buffers ready, prev free
        if (wave == 0 && kt + 1 < KT) {
            const unsigned short* An = Atile + (size_t)(kt + 1) * TILE_K;
            const unsigned short* Bn = Btile + (size_t)(kt + 1) * TILE_K;
            tdm_load_2d(lds_base + (unsigned)((b ^ 1) * A_BUF_ELEMS) * 2u, An, BM, K);
            tdm_load_2d(lds_base + (unsigned)(2 * A_BUF_ELEMS + (b ^ 1) * B_BUF_ELEMS) * 2u,
                        Bn, BN, K);
        }
        const unsigned short* sA = smem + b * A_BUF_ELEMS;
        const unsigned short* sB = smem + 2 * A_BUF_ELEMS + b * B_BUF_ELEMS;

        // Issue ALL fragment loads first, then the WMMA burst.
        FragAB a, bf[4];
        const unsigned short* ap = sA + (wm * 16 + lidx) * TILE_K + hf * 16;
        a.q[0] = *reinterpret_cast<const uint4*>(ap);
        a.q[1] = *reinterpret_cast<const uint4*>(ap + 8);
#pragma unroll
        for (int nt = 0; nt < 4; ++nt) {
            const unsigned short* bp = sB + (wn * 64 + nt * 16 + lidx) * TILE_K + hf * 16;
            bf[nt].q[0] = *reinterpret_cast<const uint4*>(bp);
            bf[nt].q[1] = *reinterpret_cast<const uint4*>(bp + 8);
        }
#pragma unroll
        for (int nt = 0; nt < 4; ++nt)
            acc[nt] = wmma_bf16(a.v, bf[nt].v, acc[nt]);
    }
#else
    // Fallback: cooperative global->LDS fill, single buffer.
    for (int kt = 0; kt < KT; ++kt) {
        {
            int row = threadIdx.x >> 2, c = threadIdx.x & 3;
            const uint4 v = *reinterpret_cast<const uint4*>(
                Atile + (size_t)row * K + kt * TILE_K + c * 8);
            *reinterpret_cast<uint4*>(smem + row * TILE_K + c * 8) = v;
        }
#pragma unroll
        for (int r = 0; r < 2; ++r) {
            int chunk = threadIdx.x + r * 256;
            int row = chunk >> 2, c = chunk & 3;
            const uint4 v = *reinterpret_cast<const uint4*>(
                Btile + (size_t)row * K + kt * TILE_K + c * 8);
            *reinterpret_cast<uint4*>(smem + 2 * A_BUF_ELEMS + row * TILE_K + c * 8) = v;
        }
        __syncthreads();
        const unsigned short* sA = smem;
        const unsigned short* sB = smem + 2 * A_BUF_ELEMS;
        FragAB a, bf[4];
        const unsigned short* ap = sA + (wm * 16 + lidx) * TILE_K + hf * 16;
        a.q[0] = *reinterpret_cast<const uint4*>(ap);
        a.q[1] = *reinterpret_cast<const uint4*>(ap + 8);
#pragma unroll
        for (int nt = 0; nt < 4; ++nt) {
            const unsigned short* bp = sB + (wn * 64 + nt * 16 + lidx) * TILE_K + hf * 16;
            bf[nt].q[0] = *reinterpret_cast<const uint4*>(bp);
            bf[nt].q[1] = *reinterpret_cast<const uint4*>(bp + 8);
        }
#pragma unroll
        for (int nt = 0; nt < 4; ++nt)
            acc[nt] = wmma_bf16(a.v, bf[nt].v, acc[nt]);
        __syncthreads();
    }
#endif

#pragma unroll
    for (int nt = 0; nt < 4; ++nt) {
#pragma unroll
        for (int v = 0; v < 8; ++v) {
            int row = m_base + v + hf * 8;
            int col = n_base + nt * 16 + lidx;
            C[(size_t)row * Nd + col] = f2bf(acc[nt][v]);
        }
    }
}

// ---------------- per-expert GEMM: y[tok,u] = sum_d xr[tok,d] * W_low[e,u,d] --
// grid = (NEXP, DU/64); block = 8 waves, each wave owns a 16-row tile strip.

__global__ __launch_bounds__(256)
void expert_gemm_kernel(const unsigned short* __restrict__ xr,
                        const unsigned short* __restrict__ W_low,
                        const int* __restrict__ sorted,
                        const unsigned int* __restrict__ offsets,
                        const unsigned int* __restrict__ counts,
                        unsigned short* __restrict__ y) {
    const int e = blockIdx.x;
    const unsigned int cnt = counts[e];
    if (cnt == 0u) return;
    const unsigned int off = offsets[e];

    const int lane = threadIdx.x & 31;
    const int wave = threadIdx.x >> 5;
    const int hf   = lane >> 4;
    const int lidx = lane & 15;
    const int n_base = blockIdx.y * 64;
    const unsigned short* Bs = W_low + (size_t)e * DU * DD;

    for (unsigned int t = (unsigned int)wave; t * 16u < cnt; t += 8u) {
        unsigned int r  = t * 16u + (unsigned int)lidx;
        unsigned int rc = (r < cnt) ? r : (cnt - 1u);   // clamp: EXEC stays full
        const unsigned short* arow = xr + (size_t)sorted[off + rc] * DD;

        f32x8 zero = {0.f, 0.f, 0.f, 0.f, 0.f, 0.f, 0.f, 0.f};
        f32x8 acc[4] = {zero, zero, zero, zero};

#pragma unroll
        for (int k = 0; k < DD; k += 32) {
            FragAB a, bf[4];
            a.q[0] = *reinterpret_cast<const uint4*>(arow + k + hf * 16);
            a.q[1] = *reinterpret_cast<const uint4*>(arow + k + hf * 16 + 8);
#pragma unroll
            for (int nt = 0; nt < 4; ++nt) {
                const unsigned short* bp =
                    Bs + (size_t)(n_base + nt * 16 + lidx) * DD + k + hf * 16;
                bf[nt].q[0] = *reinterpret_cast<const uint4*>(bp);
                bf[nt].q[1] = *reinterpret_cast<const uint4*>(bp + 8);
            }
#pragma unroll
            for (int nt = 0; nt < 4; ++nt)
                acc[nt] = wmma_bf16(a.v, bf[nt].v, acc[nt]);
        }

#pragma unroll
        for (int v = 0; v < 8; ++v) {
            unsigned int rr = t * 16u + (unsigned int)(v + hf * 8);
            if (rr < cnt) {
                int tok = sorted[off + rr];      // scatter back to original order
#pragma unroll
                for (int nt = 0; nt < 4; ++nt)
                    y[(size_t)tok * DU + n_base + nt * 16 + lidx] = f2bf(acc[nt][v]);
            }
        }
    }
}

// -------------------------------- launch --------------------------------

extern "C" void kernel_launch(void* const* d_in, const int* in_sizes, int n_in,
                              void* d_out, int out_size, void* d_ws, size_t ws_size,
                              hipStream_t stream) {
    const unsigned short* x     = (const unsigned short*)d_in[0]; // [N, IN_DIM] bf16
    const unsigned short* U_in  = (const unsigned short*)d_in[1]; // [IN_DIM, DD] bf16
    const unsigned short* U_out = (const unsigned short*)d_in[2]; // [OUT_DIM, DU] bf16
    const unsigned short* W_low = (const unsigned short*)d_in[3]; // [E, DU, DD] bf16
    const int*            eidx  = (const int*)d_in[4];            // [N] int32
    unsigned short* out = (unsigned short*)d_out;                 // [N, OUT_DIM] bf16

    char* ws = (char*)d_ws;
    size_t o = 0;
    auto take = [&](size_t bytes) { void* p = ws + o; o = (o + bytes + 255) & ~(size_t)255; return p; };

    unsigned short* U_in_T = (unsigned short*)take((size_t)DD * IN_DIM * 2);
    unsigned short* xr     = (unsigned short*)take((size_t)N_TOK * DD * 2);
    unsigned short* y      = (unsigned short*)take((size_t)N_TOK * DU * 2);
    unsigned int*   counts  = (unsigned int*)take(NEXP * 4);
    unsigned int*   offsets = (unsigned int*)take(NEXP * 4);
    unsigned int*   cursors = (unsigned int*)take(NEXP * 4);
    int*            sorted  = (int*)take(N_TOK * 4);

    // 1) group tokens by expert
    zero_counts_kernel<<<1, 64, 0, stream>>>(counts, cursors);
    hist_kernel<<<N_TOK / 256, 256, 0, stream>>>(eidx, counts);
    prefix_kernel<<<1, 64, 0, stream>>>(counts, offsets, cursors);
    scatter_kernel<<<N_TOK / 256, 256, 0, stream>>>(eidx, cursors, sorted);

    // 2) transpose U_in -> [DD, IN_DIM] so GEMM1 B-fragments load contiguously
    transpose_uin_kernel<<<(DD * IN_DIM) / 256, 256, 0, stream>>>(U_in, U_in_T);

    // 3) xr = x @ U_in            (M=8192, Nd=256, K=4096)
    gemm_bt_kernel<<<dim3(N_TOK / BM, DD / BN), 256, 0, stream>>>(
        x, U_in_T, xr, N_TOK, DD, IN_DIM);

    // 4) y[tok] = W_low[e(tok)] @ xr[tok]   (per-expert WMMA GEMM)
    expert_gemm_kernel<<<dim3(NEXP, DU / 64), 256, 0, stream>>>(
        xr, W_low, sorted, offsets, counts, y);

    // 5) out = y @ U_out^T        (M=8192, Nd=4096, K=256)
    gemm_bt_kernel<<<dim3(N_TOK / BM, OUT_DIM / BN), 256, 0, stream>>>(
        y, U_out, out, N_TOK, OUT_DIM, DD);
}
